// DepthAwareFE_57174604644721
// MI455X (gfx1250) — matrix operand order repaired and checked by reading, not compiled, not run on hardware
//
#include <hip/hip_runtime.h>
#include <hip/hip_bf16.h>

// CDNA5 (gfx1250) wave32 pipeline for DepthAwareFE.
// - 155 GFLOP fused upsample+conv3x3(+BN/ReLU)+conv1x1 via v_wmma_f32_16x16x32_bf16
//   (bf16 inputs, fp32 accumulate; compute-bound -> low-precision WMMA path)
// - feat conv1x1 (8.6 GFLOP, fp32 data) via native v_wmma_f32_16x16x4_f32 with the
//   x-tile staged in LDS by global_load_async_to_lds_b128 (ASYNCcnt path)
// - attention tail collapsed algebraically into tiny GEMMs + rank-12 streaming update.

typedef __attribute__((ext_vector_type(16))) __bf16 v16bf;
typedef __attribute__((ext_vector_type(8)))  __bf16 v8bf;
typedef __attribute__((ext_vector_type(8)))  float  v8f;
typedef __attribute__((ext_vector_type(2)))  float  v2f;

#define N_BATCH 4
#define C_IN    256
#define H_IN    128
#define W_IN    128
#define PIX     (H_IN*W_IN)        // 16384
#define H2      256
#define W2      256
#define C_D1    128                // conv3x3 out channels
#define K1      (C_IN*9)           // 2304 = 9 taps * 256 ci  (tap-major packing)
#define C_DEP   96
#define NDG     12
#define BN_EPS  1e-5f

// d_out regions (float element offsets)
#define OUT_DEPTH 0
#define OUT_GUIDE (N_BATCH*C_DEP*H2*W2)            // 25165824
#define OUT_FINAL (OUT_GUIDE + N_BATCH*NDG*PIX)    // 25952256

// workspace byte offsets (256-aligned)
#define WS_PW1 0u         // 128*2304 bf16  = 589824 B (tap-major K)
#define WS_PW2 589824u    // 96*128  bf16   = 24576 B
#define WS_EN  614400u    // 4*12*256 f32   = 49152 B
#define WS_AT  663552u    // attn           = 49152 B
#define WS_M   712704u    // 4*256*12 f32   = 49152 B

// ---------------- WMMA fragment helpers ----------------

// 16-bit A 16x32 (MxK): lanes 0-15 m=lane, K 0..15;
// lanes 16-31 m=lane-16, K 16..31. Two contiguous 16B loads + shuffle.
__device__ __forceinline__ v16bf load_a_frag(const __bf16* __restrict__ arow,
                                             int kk, int lane) {
  int half = lane >> 4;
  const __bf16* p0 = arow + kk + 16 * half;
  v8bf lo = *(const v8bf*)p0;
  v8bf hi = *(const v8bf*)(p0 + 8);
  return __builtin_shufflevector(lo, hi, 0, 1, 2, 3, 4, 5, 6, 7,
                                 8, 9, 10, 11, 12, 13, 14, 15);
}

// Swizzled LDS staging so a lane's 16 B-halfs are contiguous (2x ds_load_b128).
// Logical (k, p) -> swizzled offset.
__device__ __forceinline__ int bswz(int k, int p) {
  return ((k >> 4) << 8) + (p << 4) + (k & 15);
}
// 16-bit B 32x16 (KxN): lanes 0-15 n=lane K 0..15; lanes 16-31 n=lane-16 K 16..31.
__device__ __forceinline__ v16bf load_b_frag_swz(const __bf16* Bs, int kk, int lane) {
  const __bf16* ptr = Bs + (((kk >> 4) + (lane >> 4)) << 8) + ((lane & 15) << 4);
  return *(const v16bf*)ptr;
}

// ---------------- small utility kernels ----------------

__global__ void k_pack_w1(const float* __restrict__ src, __bf16* __restrict__ dst) {
  int i = blockIdx.x * 256 + threadIdx.x;          // 128*2304
  int co = i / K1;
  int k  = i - co * K1;
  int tap = k >> 8, ci = k & 255;                  // tap-major K layout
  dst[i] = (__bf16)src[(co * 256 + ci) * 9 + tap];
}

__global__ void k_pack_bf16(const float* __restrict__ src, __bf16* __restrict__ dst, int n) {
  int i = blockIdx.x * 256 + threadIdx.x;
  if (i < n) dst[i] = (__bf16)src[i];
}

__global__ void k_zero_f32(float* __restrict__ p, int n) {
  int i = blockIdx.x * 256 + threadIdx.x;
  if (i < n) p[i] = 0.0f;
}

// ---------------- fused upsample + conv3x3 + BN + ReLU + conv1x1 ----------------
// Grid: 4(n) * 256(y) * 16(x-tiles of 16) blocks of 256 threads (8 waves).
// tap-major K: bilinear coefficients are loop-invariant across the 8 chunks of
// each tap; only the channel plane changes in the inner loop.
__global__ __launch_bounds__(256) void k_conv_main(
    const float* __restrict__ x, const __bf16* __restrict__ pw1,
    const float* __restrict__ b_do1, const float* __restrict__ g1,
    const float* __restrict__ be1, const float* __restrict__ m1,
    const float* __restrict__ v1,
    const __bf16* __restrict__ pw2, const float* __restrict__ b_do2,
    float* __restrict__ depth) {
  __shared__ __bf16 Bsm[32 * 16];      // swizzled chunk tile
  __shared__ __bf16 Dsm[C_D1 * 16];    // swizzled d tile (stage-2 B operand)

  int t = threadIdx.x, lane = t & 31, wave = t >> 5;
  int b = blockIdx.x;
  int n = b >> 12;
  int rem = b & 4095;
  int y = rem >> 4;
  int xbase = (rem & 15) << 4;
  const float* xn = x + (size_t)n * C_IN * PIX;

  // this thread fills (kl, p0) and (kl, p0+1) of each chunk
  int kl = t >> 3;                 // 0..31
  int p0 = (t << 1) & 15;          // even
  const __bf16* arow = pw1 + (size_t)(wave * 16 + (lane & 15)) * K1;

  const float sc = 127.0f / 255.0f;
  v8f acc = {};

  for (int tap = 0; tap < 9; ++tap) {
    int r = tap / 3, s = tap - r * 3;
    int uy = y + r - 1;
    float sy = (float)uy * sc;
    int   y0 = (int)sy;            // trunc-to-zero: safe for uy=-1 too
    float wy = sy - (float)y0;
    int   y1 = min(y0 + 1, H_IN - 1);
    bool  vy = (unsigned)uy < (unsigned)H2;

    int ux0 = xbase + p0 + s - 1;
    int ux1 = ux0 + 1;
    float sx0 = (float)ux0 * sc, sx1 = (float)ux1 * sc;
    int   x00 = (int)sx0,        x10 = (int)sx1;
    float wx0 = sx0 - (float)x00, wx1 = sx1 - (float)x10;
    int   x01 = min(x00 + 1, W_IN - 1), x11 = min(x10 + 1, W_IN - 1);
    bool  v0 = vy && ((unsigned)ux0 < (unsigned)W2);
    bool  v1m = vy && ((unsigned)ux1 < (unsigned)W2);
    if (ux0 < 0) { x00 = 0; x01 = 0; wx0 = 0.f; }
    if (sy < 0.f) { y0 = 0; wy = 0.f; }

    for (int cc = 0; cc < 8; ++cc) {
      int kk = tap * 256 + cc * 32;
      __syncthreads();
      {
        const float* pl = xn + (size_t)(cc * 32 + kl) * PIX;
        const float* r0 = pl + y0 * W_IN;
        const float* r1 = pl + y1 * W_IN;
        float t00 = r0[x00] + (r0[x01] - r0[x00]) * wx0;
        float t01 = r1[x00] + (r1[x01] - r1[x00]) * wx0;
        float t10 = r0[x10] + (r0[x11] - r0[x10]) * wx1;
        float t11 = r1[x10] + (r1[x11] - r1[x10]) * wx1;
        float b0 = t00 + (t01 - t00) * wy;
        float b1 = t10 + (t11 - t10) * wy;
        Bsm[bswz(kl, p0)]     = (__bf16)(v0  ? b0 : 0.0f);
        Bsm[bswz(kl, p0 + 1)] = (__bf16)(v1m ? b1 : 0.0f);
      }
      __syncthreads();
      v16bf af = load_a_frag(arow, kk, lane);
      v16bf bf = load_b_frag_swz(Bsm, 0, lane);
      acc = __builtin_amdgcn_wmma_f32_16x16x32_bf16(false, af, false, bf,
                                                    (short)0, acc, false, false);
    }
  }

  // epilogue: bias + BN + ReLU -> LDS (bf16, swizzled for stage 2)
  int p = lane & 15, half = lane >> 4;
#pragma unroll
  for (int vi = 0; vi < 8; ++vi) {
    int c = wave * 16 + vi + 8 * half;            // 0..127
    float s1 = g1[c] * rsqrtf(v1[c] + BN_EPS);
    float sh = (b_do1[c] - m1[c]) * s1 + be1[c];
    float dv = fmaxf(acc[vi] * s1 + sh, 0.0f);
    Dsm[bswz(c, p)] = (__bf16)dv;
  }
  __syncthreads();

  if (wave < 6) {                                  // 6 waves cover 96 channels
    const __bf16* arow2 = pw2 + (size_t)(wave * 16 + (lane & 15)) * C_D1;
    v8f acc2 = {};
#pragma unroll
    for (int kk = 0; kk < C_D1; kk += 32) {
      v16bf af = load_a_frag(arow2, kk, lane);
      v16bf bf = load_b_frag_swz(Dsm, kk, lane);
      acc2 = __builtin_amdgcn_wmma_f32_16x16x32_bf16(false, af, false, bf,
                                                     (short)0, acc2, false, false);
    }
#pragma unroll
    for (int vi = 0; vi < 8; ++vi) {
      int co = wave * 16 + vi + 8 * half;          // 0..95
      depth[(((size_t)n * C_DEP + co) * H2 + y) * W2 + xbase + p] = acc2[vi] + b_do2[co];
    }
  }
}

// ---------------- avgpool(exact 1/2 bilinear) + grouped conv3x3 ----------------
__global__ void k_guide(const float* __restrict__ depth, const float* __restrict__ w_dd,
                        const float* __restrict__ b_dd, float* __restrict__ guide) {
  int id = blockIdx.x * 256 + threadIdx.x;           // 4*12*16384
  int n = id / (NDG * PIX);
  int rem = id - n * (NDG * PIX);
  int o = rem / PIX;
  int pix = rem - o * PIX;
  int yy = pix >> 7, xx = pix & 127;
  float acc = b_dd[o];
#pragma unroll
  for (int r = 0; r < 3; ++r) {
    int iy = yy + r - 1;
    if ((unsigned)iy >= (unsigned)H_IN) continue;
#pragma unroll
    for (int s = 0; s < 3; ++s) {
      int ix = xx + s - 1;
      if ((unsigned)ix >= (unsigned)W_IN) continue;
      for (int ic = 0; ic < 8; ++ic) {
        int c = o * 8 + ic;
        const float* dp = depth + (((size_t)n * C_DEP + c) * H2 + 2 * iy) * W2 + 2 * ix;
        float dg = 0.25f * (dp[0] + dp[1] + dp[W2] + dp[W2 + 1]);
        acc += w_dd[(o * 8 + ic) * 9 + r * 3 + s] * dg;
      }
    }
  }
  guide[id] = acc;
}

// ---------------- conv1x1(x,256->256)+BN+ReLU fused with energy reduction ----------------
// fp32 path: x-tile (256K x 16px) staged in LDS via global_load_async_to_lds_b128,
// consumed by v_wmma_f32_16x16x4_f32. Grid: 4(n)*2(cout half)*1024(px tiles) blocks.
__global__ __launch_bounds__(256) void k_feat_energy(
    const float* __restrict__ x, const float* __restrict__ wacf1,
    const float* __restrict__ g2, const float* __restrict__ be2,
    const float* __restrict__ m2, const float* __restrict__ v2,
    const float* __restrict__ qg, float* __restrict__ energy) {
  __shared__ float Xsm[C_IN * 16];   // 16 KB, k-major [256][16]
  __shared__ float Fsm[128 * 16];    // 8 KB
  __shared__ float Qs[NDG * 16];

  int t = threadIdx.x, lane = t & 31, wave = t >> 5;
  int b = blockIdx.x;
  int n = b >> 11;
  int rem = b & 2047;
  int chalf = rem >> 10;
  int pbase = (rem & 1023) << 4;
  const float* xn = x + (size_t)n * C_IN * PIX;

  // async copy the whole 256x16 fp32 tile into LDS (4x b128 per thread)
#pragma unroll
  for (int i = 0; i < 4; ++i) {
    int c4 = i * 256 + t;                   // 16-byte chunk index, 0..1023
    int k = c4 >> 2;                        // row 0..255
    int poff = (c4 & 3) << 2;               // 0,4,8,12
    const float* gp = xn + (size_t)k * PIX + pbase + poff;
    unsigned lo = (unsigned)(size_t)(&Xsm[(c4 << 2)]);
    asm volatile("global_load_async_to_lds_b128 %0, %1, off"
                 :: "v"(lo), "v"(gp) : "memory");
  }
  asm volatile("s_wait_asynccnt 0x0" ::: "memory");
  __syncthreads();

  // fp32 WMMA 16x16x4, K = 256
  int p = lane & 15, half = lane >> 4;
  int mrow = chalf * 128 + wave * 16 + (lane & 15);
  const float* wrow = wacf1 + (size_t)mrow * 256;
  v8f acc = {};
  for (int kb = 0; kb < C_IN; kb += 4) {
    v2f a = *(const v2f*)(wrow + kb + 2 * half);
    v2f bb;
    bb[0] = Xsm[(kb + 2 * half)     * 16 + p];
    bb[1] = Xsm[(kb + 2 * half + 1) * 16 + p];
    acc = __builtin_amdgcn_wmma_f32_16x16x4_f32(false, a, false, bb,
                                                (short)0, acc, false, false);
  }

#pragma unroll
  for (int vi = 0; vi < 8; ++vi) {
    int cl = wave * 16 + vi + 8 * half;           // local 0..127
    int c = chalf * 128 + cl;
    float s2 = g2[c] * rsqrtf(v2[c] + BN_EPS);
    float sh = be2[c] - m2[c] * s2;
    Fsm[cl * 16 + p] = fmaxf(acc[vi] * s2 + sh, 0.0f);
  }
  if (t < NDG * 16) {
    int d = t >> 4, pp = t & 15;
    Qs[t] = qg[((size_t)n * NDG + d) * PIX + pbase + pp];
  }
  __syncthreads();

  for (int pair = t; pair < NDG * 128; pair += 256) {
    int d = pair / 128;
    int cl = pair - d * 128;
    float s = 0.0f;
#pragma unroll
    for (int pp = 0; pp < 16; ++pp) s += Fsm[cl * 16 + pp] * Qs[d * 16 + pp];
    atomicAdd(&energy[((size_t)n * NDG + d) * 256 + chalf * 128 + cl], s);
  }
}

// ---------------- softmax over C of (rowmax - energy): attn = exp(rowmin-e)/sum ----------------
__global__ void k_softmax(const float* __restrict__ energy, float* __restrict__ attn) {
  __shared__ float red[256];
  int row = blockIdx.x;                      // 0..47 = (n,d)
  int t = threadIdx.x;
  float e = energy[row * 256 + t];
  red[t] = e;
  __syncthreads();
  for (int s = 128; s > 0; s >>= 1) {
    if (t < s) red[t] = fminf(red[t], red[t + s]);
    __syncthreads();
  }
  float mn = red[0];
  __syncthreads();
  float pv = __expf(mn - e);
  red[t] = pv;
  __syncthreads();
  for (int s = 128; s > 0; s >>= 1) {
    if (t < s) red[t] += red[t + s];
    __syncthreads();
  }
  attn[row * 256 + t] = pv / red[0];
}

// ---------------- M[n,co,d] = sum_ci w_acf2[co,ci] * attn[n,d,ci] ----------------
__global__ void k_mmat(const float* __restrict__ w2, const float* __restrict__ attn,
                       float* __restrict__ M) {
  int id = blockIdx.x * 256 + threadIdx.x;   // 4*256*12 = 12288
  int n = id / (256 * NDG);
  int rem = id - n * (256 * NDG);
  int co = rem / NDG;
  int d = rem - co * NDG;
  float s = 0.0f;
  const float* wr = w2 + (size_t)co * 256;
  const float* ar = attn + ((size_t)n * NDG + d) * 256;
  for (int ci = 0; ci < 256; ++ci) s += wr[ci] * ar[ci];
  M[id] = s;
}

// ---------------- final = x + b_acf2 + sum_d M[n,co,d] * q[n,d,p] ----------------
__global__ void k_final(const float* __restrict__ x, const float* __restrict__ b2,
                        const float* __restrict__ M, const float* __restrict__ qg,
                        float* __restrict__ outp) {
  size_t id = (size_t)blockIdx.x * 256 + threadIdx.x;  // 4*256*16384
  int n = (int)(id >> 22);
  int rem = (int)(id & 4194303);
  int co = rem >> 14;
  int p = rem & 16383;
  const float* Mr = M + ((size_t)n * 256 + co) * NDG;
  const float* qr = qg + (size_t)n * NDG * PIX + p;
  float acc = x[id] + b2[co];
#pragma unroll
  for (int d = 0; d < NDG; ++d) acc += Mr[d] * qr[(size_t)d * PIX];
  outp[id] = acc;
}

// ---------------- launch ----------------

extern "C" void kernel_launch(void* const* d_in, const int* in_sizes, int n_in,
                              void* d_out, int out_size, void* d_ws, size_t ws_size,
                              hipStream_t stream) {
  const float* x     = (const float*)d_in[0];
  const float* w_do1 = (const float*)d_in[1];
  const float* b_do1 = (const float*)d_in[2];
  const float* g1    = (const float*)d_in[3];
  const float* be1   = (const float*)d_in[4];
  const float* m1    = (const float*)d_in[5];
  const float* v1    = (const float*)d_in[6];
  const float* w_do2 = (const float*)d_in[7];
  const float* b_do2 = (const float*)d_in[8];
  const float* w_dd  = (const float*)d_in[9];
  const float* b_dd  = (const float*)d_in[10];
  const float* wacf1 = (const float*)d_in[11];
  const float* g2    = (const float*)d_in[12];
  const float* be2   = (const float*)d_in[13];
  const float* m2    = (const float*)d_in[14];
  const float* v2    = (const float*)d_in[15];
  const float* wacf2 = (const float*)d_in[16];
  const float* bacf2 = (const float*)d_in[17];

  float* out   = (float*)d_out;
  float* depth = out + OUT_DEPTH;
  float* guide = out + OUT_GUIDE;
  float* fin   = out + OUT_FINAL;

  char* ws = (char*)d_ws;
  __bf16* pw1    = (__bf16*)(ws + WS_PW1);
  __bf16* pw2    = (__bf16*)(ws + WS_PW2);
  float*  energy = (float*)(ws + WS_EN);
  float*  attn   = (float*)(ws + WS_AT);
  float*  M      = (float*)(ws + WS_M);

  k_pack_w1<<<(C_D1 * K1) / 256, 256, 0, stream>>>(w_do1, pw1);
  k_pack_bf16<<<(C_DEP * C_D1 + 255) / 256, 256, 0, stream>>>(w_do2, pw2, C_DEP * C_D1);
  k_zero_f32<<<(N_BATCH * NDG * 256 + 255) / 256, 256, 0, stream>>>(energy, N_BATCH * NDG * 256);

  k_conv_main<<<N_BATCH * H2 * (W2 / 16), 256, 0, stream>>>(
      x, pw1, b_do1, g1, be1, m1, v1, pw2, b_do2, depth);

  k_guide<<<(N_BATCH * NDG * PIX) / 256, 256, 0, stream>>>(depth, w_dd, b_dd, guide);

  k_feat_energy<<<N_BATCH * 2 * (PIX / 16), 256, 0, stream>>>(
      x, wacf1, g2, be2, m2, v2, guide, energy);

  k_softmax<<<N_BATCH * NDG, 256, 0, stream>>>(energy, attn);
  k_mmat<<<(N_BATCH * 256 * NDG) / 256, 256, 0, stream>>>(wacf2, attn, M);
  k_final<<<(N_BATCH * 256 * PIX) / 256, 256, 0, stream>>>(x, bacf2, M, guide, fin);
}
